// Dynamic_System_13134009991229
// MI455X (gfx1250) — matrix operand order repaired and checked by reading, not compiled, not run on hardware
//
#include <hip/hip_runtime.h>
#include <hip/hip_bf16.h>

// ---------------- CDNA5 WMMA types ----------------
typedef __attribute__((ext_vector_type(16))) __bf16        v16bf;
typedef __attribute__((ext_vector_type(8)))  float         v8f;
typedef __attribute__((ext_vector_type(4)))  unsigned int  u32x4;

union Frag {
    v16bf  v;
    u32x4  q[2];
    __bf16 h[16];
};

__device__ inline v8f wmma_bf16(v16bf a, v16bf b, v8f c) {
    return __builtin_amdgcn_wmma_f32_16x16x32_bf16(
        /*neg_a=*/false, a, /*neg_b=*/false, b,
        /*c_mod=*/(short)0, c, /*reuse_a=*/false, /*reuse_b=*/false);
}

// ---------------- problem constants ----------------
#define NB      131072
#define HID     256
#define SPW     16          // samples per wave (WMMA M)
#define WAVES   8
#define WGS     (SPW*WAVES) // 128 samples per workgroup
#define ASTR    528         // activation row stride in bytes (264 bf16, bank-friendly)

// ---------------- LDS layout (bytes) ----------------
#define ACTB      (WGS*ASTR)        // 67584
#define OFF_H1    0
#define OFF_G2    (OFF_H1  + ACTB)  // g2, later reused for dg2_j
#define OFF_M2    (OFF_G2  + ACTB)
#define OFF_G1H   (OFF_M2  + ACTB)
#define OFF_X     (OFF_G1H + ACTB)        // 128 x 16 f32 staged inputs
#define OFF_W1    (OFF_X   + WGS*16*4)    // 10 x 256 f32
#define OFF_B1    (OFF_W1  + 10*256*4)
#define OFF_B2    (OFF_B1  + 256*4)
#define OFF_W3    (OFF_B2  + 256*4)
#define OFF_BW1   (OFF_W3  + 256*4)       // 9 x 256 f32
#define OFF_BB1   (OFF_BW1 + 9*256*4)
#define OFF_BW2   (OFF_BB1 + 256*4)       // 256 x 3 f32
#define OFF_HC    (OFF_BW2 + 256*3*4)     // 128 x 9 f32  (H columns)
#define OFF_CC    (OFF_HC  + WGS*9*4)     // 128 x 9 f32  (C columns)
#define OFF_DLT   (OFF_CC  + WGS*9*4)     // 128 x 4 f32  (dL/dtheta)
#define OFF_END   (OFF_DLT + WGS*4*4)     // ~316 KB total (<= 320 KB WGP LDS)

// ---------------- prep kernel: bf16 weight images into d_ws ----------------
// wimg layout (bf16 elements):
//   [0*65536]  W2   row-major [k][n]          (B-source for X @ W2^T)
//   [1*65536]  W2^T row-major [n][k]          (B-source for X @ W2)
//   [(2+j)*65536] (diag(W1_j) * W2)^T [n][k]  (B-source for tangent GEMM, j=0..2)
__global__ void dyn_prep(const float* __restrict__ LW1,
                         const float* __restrict__ LW2,
                         __bf16* __restrict__ wimg) {
    int idx = blockIdx.x * 256 + threadIdx.x;   // 0..65535
    int k = idx >> 8, n = idx & 255;
    float w = LW2[idx];                          // LW2[k][n]
    wimg[idx]                = (__bf16)w;        // W2 image
    wimg[65536 + n*256 + k]  = (__bf16)w;        // W2^T image
#pragma unroll
    for (int j = 0; j < 3; ++j)
        wimg[(2+j)*65536 + n*256 + k] = (__bf16)(w * LW1[j*256 + k]);
}

// K=256 WMMA accumulation: A fragments from an LDS activation row (bf16,
// optionally transformed to t1 = 1-h^2 on the fly), B fragments from a
// contiguous row of a (transposed) global weight image.
template <bool T1>
__device__ inline v8f gemm_k256(const char* aptr, const __bf16* brow, int hsel) {
    v8f acc = {0.f,0.f,0.f,0.f,0.f,0.f,0.f,0.f};
#pragma unroll
    for (int ks = 0; ks < 8; ++ks) {
        Frag A, Bf;
        int ka = ks*64 + hsel*16;                 // byte offset of first A chunk
        A.q[0] = *(const u32x4*)(aptr + ka);
        A.q[1] = *(const u32x4*)(aptr + ka + 32);
        if (T1) {
#pragma unroll
            for (int i = 0; i < 16; ++i) {
                float h = (float)A.h[i];
                A.h[i] = (__bf16)(1.0f - h*h);
            }
        }
        const char* bb = (const char*)brow + ks*64 + hsel*16;
        Bf.q[0] = *(const u32x4*)(bb);
        Bf.q[1] = *(const u32x4*)(bb + 32);
        acc = wmma_bf16(A.v, Bf.v, acc);
    }
    return acc;
}

__global__ __launch_bounds__(256)
void dyn_main(const float* __restrict__ theta_dot, const float* __restrict__ s_dot,
              const float* __restrict__ theta,     const float* __restrict__ z,
              const float* __restrict__ s,         const float* __restrict__ s_Ddot,
              const float* __restrict__ tau,
              const float* __restrict__ LW1, const float* __restrict__ Lb1,
              const float* __restrict__ Lb2, const float* __restrict__ LW3,
              const float* __restrict__ BW1, const float* __restrict__ Bb1,
              const float* __restrict__ BW2, const float* __restrict__ Bb2,
              const __bf16* __restrict__ wimg,
              float* __restrict__ out) {
    extern __shared__ char smem[];
    const int tid    = threadIdx.x;
    const int wgBase = blockIdx.x * WGS;

    float* X    = (float*)(smem + OFF_X);
    float* W1s  = (float*)(smem + OFF_W1);
    float* B1s  = (float*)(smem + OFF_B1);
    float* B2s  = (float*)(smem + OFF_B2);
    float* W3s  = (float*)(smem + OFF_W3);
    float* BW1s = (float*)(smem + OFF_BW1);
    float* BB1s = (float*)(smem + OFF_BB1);
    float* BW2s = (float*)(smem + OFF_BW2);
    float* HC   = (float*)(smem + OFF_HC);
    float* CC   = (float*)(smem + OFF_CC);
    float* DLT  = (float*)(smem + OFF_DLT);

    // ---- cooperative staging ----
    for (int i = tid; i < WGS*16; i += 256) {
        int r = i >> 4, c = i & 15, gs = wgBase + r;
        float v;
        if      (c <  3) v = theta_dot[gs*3 + c];
        else if (c <  6) v = s_dot   [gs*3 + (c-3)];
        else if (c <  9) v = theta   [gs*3 + (c-6)];
        else if (c == 9) v = z       [gs];
        else if (c < 13) v = s       [gs*3 + (c-10)];
        else             v = s_Ddot  [gs*3 + (c-13)];
        X[i] = v;
    }
    for (int i = tid; i < 2560; i += 256) W1s[i] = LW1[i];
    for (int i = tid; i < 2304; i += 256) BW1s[i] = BW1[i];
    for (int i = tid; i <  768; i += 256) BW2s[i] = BW2[i];
    if (tid < 256) { B1s[tid] = Lb1[tid]; B2s[tid] = Lb2[tid];
                     W3s[tid] = LW3[tid]; BB1s[tid] = Bb1[tid]; }
    __syncthreads();

    const int lane = tid & 31;
    const int m0   = (tid >> 5) * SPW;  // this wave's local sample base
    const int hsel = lane >> 4;
    const int nl   = lane & 15;

    char* h1b = smem + OFF_H1;
    char* g2b = smem + OFF_G2;
    char* m2b = smem + OFF_M2;
    char* ghb = smem + OFF_G1H;

    const __bf16* W2img  = wimg;
    const __bf16* W2Timg = wimg + 65536;

    // ---- phase 1: a1 = x @ W1 + b1 ; h1 = tanh ----
    for (int e = 0; e < 128; ++e) {
        int idx = e*32 + lane;
        int m = idx >> 8, n = idx & 255;
        int row = m0 + m;
        float a = B1s[n];
#pragma unroll
        for (int i = 0; i < 10; ++i) a += X[row*16 + i] * W1s[i*256 + n];
        *(__bf16*)(h1b + row*ASTR + n*2) = (__bf16)tanhf(a);
    }

    const char* aRowH1 = h1b + (m0 + nl)*ASTR;
    const char* aRowG2 = g2b + (m0 + nl)*ASTR;

    // ---- phase 2: a2 = h1 @ W2  ->  g2 = W3*t2 ; m2 = -2*h2*t2*W3 ----
    for (int nt = 0; nt < 16; ++nt) {
        int n = nt*16 + nl;
        v8f acc = gemm_k256<false>(aRowH1, W2Timg + n*256, hsel);
#pragma unroll
        for (int r = 0; r < 8; ++r) {
            int lrow = m0 + r + 8*hsel;
            float h2 = tanhf(acc[r] + B2s[n]);
            float t2 = 1.f - h2*h2;
            float w3 = W3s[n];
            *(__bf16*)(g2b + lrow*ASTR + n*2) = (__bf16)(w3 * t2);
            *(__bf16*)(m2b + lrow*ASTR + n*2) = (__bf16)(-2.f * h2 * t2 * w3);
        }
    }

    // ---- phase 3: g1h = g2 @ W2^T ; store g1h ; accumulate dL/dtheta ----
    float pTh[24];
#pragma unroll
    for (int t = 0; t < 24; ++t) pTh[t] = 0.f;
    for (int nt = 0; nt < 16; ++nt) {
        int n = nt*16 + nl;
        v8f acc = gemm_k256<false>(aRowG2, W2img + n*256, hsel);
#pragma unroll
        for (int r = 0; r < 8; ++r) {
            int lrow = m0 + r + 8*hsel;
            float gh = acc[r];
            float hv = (float)(*(const __bf16*)(h1b + lrow*ASTR + n*2));
            float t1 = 1.f - hv*hv;
            *(__bf16*)(ghb + lrow*ASTR + n*2) = (__bf16)gh;
            float g1 = gh * t1;
#pragma unroll
            for (int c = 0; c < 3; ++c) pTh[c*8 + r] += g1 * W1s[(6+c)*256 + n];
        }
    }
#pragma unroll
    for (int t = 0; t < 24; ++t)
        for (int msk = 1; msk < 16; msk <<= 1)
            pTh[t] += __shfl_xor(pTh[t], msk, 32);
    if (nl < 8) {
        int row = m0 + 8*hsel + nl;
#pragma unroll
        for (int c = 0; c < 3; ++c) DLT[row*4 + c] = pTh[c*8 + nl];
    }

    // ---- phase 4: three tangent passes (columns of H and C) ----
    for (int j = 0; j < 3; ++j) {
        const __bf16* WTj = wimg + (2+j)*65536;
        // 4a: da2_j = t1 @ (diag(W1_j) W2) ; dg2_j = m2 * da2_j  -> g2 buffer
        for (int nt = 0; nt < 16; ++nt) {
            int n = nt*16 + nl;
            v8f acc = gemm_k256<true>(aRowH1, WTj + n*256, hsel);
#pragma unroll
            for (int r = 0; r < 8; ++r) {
                int lrow = m0 + r + 8*hsel;
                float m2v = (float)(*(const __bf16*)(m2b + lrow*ASTR + n*2));
                *(__bf16*)(g2b + lrow*ASTR + n*2) = (__bf16)(m2v * acc[r]);
            }
        }
        // 4b: dg1h_j = dg2_j @ W2^T ; dg1_j ; project onto W1 rows {0..2,6..8}
        float pJ[48];
#pragma unroll
        for (int t = 0; t < 48; ++t) pJ[t] = 0.f;
        for (int nt = 0; nt < 16; ++nt) {
            int n = nt*16 + nl;
            v8f acc = gemm_k256<false>(aRowG2, W2img + n*256, hsel);
            float w1jn = W1s[j*256 + n];
#pragma unroll
            for (int r = 0; r < 8; ++r) {
                int lrow = m0 + r + 8*hsel;
                float hv = (float)(*(const __bf16*)(h1b + lrow*ASTR + n*2));
                float t1 = 1.f - hv*hv;
                float gh = (float)(*(const __bf16*)(ghb + lrow*ASTR + n*2));
                float dg1 = t1*acc[r] - 2.f*gh*hv*t1*w1jn;
#pragma unroll
                for (int c = 0; c < 3; ++c) {
                    pJ[c*8 + r]     += dg1 * W1s[c*256 + n];       // H column j
                    pJ[(3+c)*8 + r] += dg1 * W1s[(6+c)*256 + n];   // C column j
                }
            }
        }
#pragma unroll
        for (int t = 0; t < 48; ++t)
            for (int msk = 1; msk < 16; msk <<= 1)
                pJ[t] += __shfl_xor(pJ[t], msk, 32);
        if (nl < 8) {
            int row = m0 + 8*hsel + nl;
#pragma unroll
            for (int c = 0; c < 3; ++c) {
                HC[row*9 + j*3 + c] = pJ[c*8 + nl];
                CC[row*9 + j*3 + c] = pJ[(3+c)*8 + nl];
            }
        }
    }

    // ---- phase 5: B_NN (two lanes per sample) + 3x3 solve ----
    {
        int row = m0 + nl;
        float pb0 = 0.f, pb1 = 0.f, pb2 = 0.f;
        for (int e = 0; e < 128; ++e) {
            int n = e*2 + hsel;
            float a = BB1s[n];
#pragma unroll
            for (int i = 0; i < 9; ++i) {
                int slot = (i < 3) ? (6+i) : (10 + (i-3)); // [theta, s, s_Ddot]
                a += X[row*16 + slot] * BW1s[i*256 + n];
            }
            float hb = tanhf(a);
            pb0 += hb * BW2s[n*3+0];
            pb1 += hb * BW2s[n*3+1];
            pb2 += hb * BW2s[n*3+2];
        }
        pb0 += __shfl_xor(pb0, 16, 32);
        pb1 += __shfl_xor(pb1, 16, 32);
        pb2 += __shfl_xor(pb2, 16, 32);
        if (hsel == 0) {
            int gs = wgBase + row;
            float td0 = X[row*16+0], td1 = X[row*16+1], td2 = X[row*16+2];
            float r0 = tau[gs*3+0] + pb0 + Bb2[0] + DLT[row*4+0];
            float r1 = tau[gs*3+1] + pb1 + Bb2[1] + DLT[row*4+1];
            float r2 = tau[gs*3+2] + pb2 + Bb2[2] + DLT[row*4+2];
            // rhs -= C @ theta_dot   (C[i][k] = CC[row*9 + k*3 + i])
            r0 -= CC[row*9+0]*td0 + CC[row*9+3]*td1 + CC[row*9+6]*td2;
            r1 -= CC[row*9+1]*td0 + CC[row*9+4]*td1 + CC[row*9+7]*td2;
            r2 -= CC[row*9+2]*td0 + CC[row*9+5]*td1 + CC[row*9+8]*td2;
            float a00=HC[row*9+0], a01=HC[row*9+3], a02=HC[row*9+6];
            float a10=HC[row*9+1], a11=HC[row*9+4], a12=HC[row*9+7];
            float a20=HC[row*9+2], a21=HC[row*9+5], a22=HC[row*9+8];
            float c00 =  a11*a22 - a12*a21;
            float c01 = -(a10*a22 - a12*a20);
            float c02 =  a10*a21 - a11*a20;
            float c10 = -(a01*a22 - a02*a21);
            float c11 =  a00*a22 - a02*a20;
            float c12 = -(a00*a21 - a01*a20);
            float c20 =  a01*a12 - a02*a11;
            float c21 = -(a00*a12 - a02*a10);
            float c22 =  a00*a11 - a01*a10;
            float det = a00*c00 + a01*c01 + a02*c02;
            float inv = 1.f / det;
            out[gs*3+0] = (c00*r0 + c10*r1 + c20*r2) * inv;
            out[gs*3+1] = (c01*r0 + c11*r1 + c21*r2) * inv;
            out[gs*3+2] = (c02*r0 + c12*r1 + c22*r2) * inv;
        }
    }
}

extern "C" void kernel_launch(void* const* d_in, const int* in_sizes, int n_in,
                              void* d_out, int out_size, void* d_ws, size_t ws_size,
                              hipStream_t stream) {
    const float* theta_dot = (const float*)d_in[0];
    const float* s_dot     = (const float*)d_in[1];
    const float* theta     = (const float*)d_in[2];
    const float* z         = (const float*)d_in[3];
    const float* s         = (const float*)d_in[4];
    const float* s_Ddot    = (const float*)d_in[5];
    const float* tau       = (const float*)d_in[6];
    const float* LW1       = (const float*)d_in[7];
    const float* Lb1       = (const float*)d_in[8];
    const float* LW2       = (const float*)d_in[9];
    const float* Lb2       = (const float*)d_in[10];
    const float* LW3       = (const float*)d_in[11];
    // d_in[12] = Lb3 (unused: gradients do not involve the last bias)
    const float* BW1       = (const float*)d_in[13];
    const float* Bb1       = (const float*)d_in[14];
    const float* BW2       = (const float*)d_in[15];
    const float* Bb2       = (const float*)d_in[16];
    float* out = (float*)d_out;

    __bf16* wimg = (__bf16*)d_ws;   // 5 x 65536 bf16 = 640 KB of scratch

    dyn_prep<<<256, 256, 0, stream>>>(LW1, LW2, wimg);
    dyn_main<<<NB / WGS, 256, OFF_END, stream>>>(
        theta_dot, s_dot, theta, z, s, s_Ddot, tau,
        LW1, Lb1, Lb2, LW3, BW1, Bb1, BW2, Bb2, wimg, out);
}
